// FullBatchCrossEntropyLoss_31705448579423
// MI455X (gfx1250) — matrix operand order repaired and checked by reading, not compiled, not run on hardware
//
#include <hip/hip_runtime.h>
#include <hip/hip_bf16.h>
#include <math.h>

// ---------------------------------------------------------------------------
// FullBatchCrossEntropyLoss on MI455X (gfx1250, wave32)
//   logits = l2norm(gather(q)) @ l2norm(keys)^T   (2048 x 100000, D=128)
//   out    = mean( logsumexp(logits, axis=-1) - logits[n, label[n]] )
// Unit-vector operands => |logit| <= 1 => exp() needs no max shift, so the
// logsumexp is a single-pass streaming sum(exp) merged via fp32 atomicAdd.
// GEMM runs in bf16 WMMA (f32 accum); target logit computed in exact fp32.
//
// GEMM kernel: each wave owns TWO 16-row A tiles in VGPRs (2x B reuse) and
// software-pipelines (double-buffers) the B tile loads against the 8-WMMA
// chain so v_wmma never waits on same-iteration loads.
// ---------------------------------------------------------------------------

typedef __attribute__((ext_vector_type(16))) __bf16 v16bf;
typedef __attribute__((ext_vector_type(8)))  float  v8f;

#define DIM 128   // embedding dim (reference)
#define SEQ 512   // sequence length (reference: (B=8, S=512, D=128))

// ---------------- small helpers ----------------

__device__ __forceinline__ float block128_reduce(float v) {
  __shared__ float sm[128];
  sm[threadIdx.x] = v;
  __syncthreads();
#pragma unroll
  for (int off = 64; off > 0; off >>= 1) {
    if ((int)threadIdx.x < off) sm[threadIdx.x] += sm[threadIdx.x + off];
    __syncthreads();
  }
  float r = sm[0];
  __syncthreads();
  return r;
}

__global__ void zero_kernel(float* __restrict__ p, int n) {
  int i = blockIdx.x * blockDim.x + threadIdx.x;
  if (i < n) p[i] = 0.0f;
}

// ---------------- pre-pass: gather + normalize queries (fp32) ----------------
__global__ void gather_norm_q_kernel(const float* __restrict__ qe,
                                     const long long* __restrict__ loc,
                                     float* __restrict__ qf) {
  const int n = blockIdx.x;
  const int d = threadIdx.x;
  const long long b = loc[2 * n + 0];
  const long long s = loc[2 * n + 1];
  const float v = qe[((size_t)b * SEQ + (size_t)s) * DIM + d];
  const float ss = block128_reduce(v * v);
  const float inv = 1.0f / fmaxf(sqrtf(ss), 1e-12f);
  qf[(size_t)n * DIM + d] = v * inv;
}

// per-key inverse L2 norm; one block (128 threads) per key
__global__ void knorm_kernel(const float* __restrict__ ke,
                             float* __restrict__ kinv) {
  const int k = blockIdx.x;
  const int d = threadIdx.x;
  const float v = ke[(size_t)k * DIM + d];
  const float ss = block128_reduce(v * v);
  if (d == 0) kinv[k] = 1.0f / fmaxf(sqrtf(ss), 1e-12f);
}

// ---------------- packing into WMMA per-lane layouts ----------------
// A (16x32 bf16, lane=M, K striped per ISA table):
//   lanes 0-15 : M=lane,    VGPR v holds K = (v<4?0:16)  + 2*(v&3) + h
//   lanes 16-31: M=lane-16, VGPR v holds K = (v<4?8:24)  + 2*(v&3) + h
// blockIdx.x = rowTile*4 + kblock; 512 threads: lane=tid>>4, e=tid&15
__global__ void pack_q_kernel(const float* __restrict__ qf,
                              __bf16* __restrict__ Apack, int N) {
  const int tid  = threadIdx.x;
  const int lane = tid >> 4;
  const int e    = tid & 15;
  const int rt   = blockIdx.x >> 2;
  const int kb   = blockIdx.x & 3;
  const int v    = e >> 1, h = e & 1;
  int kp = ((v < 4) ? 0 : 16) + ((v & 3) << 1) + h;
  if (lane >= 16) kp += 8;
  const int M   = lane & 15;
  const int n   = rt * 16 + M;
  const int dim = kb * 32 + kp;
  const float val = (n < N) ? qf[(size_t)n * DIM + dim] : 0.0f;
  Apack[(size_t)blockIdx.x * 512 + tid] = (__bf16)val;
}

// B (32x16 bf16, lane=N, K striped like A): lane l holds column (key) N=l&15,
// element e holds K = (e<8?e:e+8) + 8*(l>>4).  blockIdx.x = vTile*4 + kblock.
__global__ void pack_k_kernel(const float* __restrict__ ke,
                              const float* __restrict__ kinv,
                              __bf16* __restrict__ Bpack, int V) {
  const int tid  = threadIdx.x;
  const int lane = tid >> 4;     // key within 16-tile (lane&15), K half (lane>>4)
  const int e    = tid & 15;     // K element
  const int t    = blockIdx.x >> 2;
  const int kb   = blockIdx.x & 3;
  const int key  = t * 16 + (lane & 15);
  const int dim  = kb * 32 + ((e < 8) ? e : e + 8) + ((lane >> 4) << 3);
  const float val = (key < V) ? ke[(size_t)key * DIM + dim] * kinv[key] : 0.0f;
  Bpack[(size_t)blockIdx.x * 512 + tid] = (__bf16)val;
}

// ---------------- exact fp32 target logits ----------------
__global__ void tgt_kernel(const float* __restrict__ qf,
                           const float* __restrict__ ke,
                           const float* __restrict__ kinv,
                           const long long* __restrict__ labels,
                           float* __restrict__ tgt) {
  const int n = blockIdx.x;
  const int d = threadIdx.x;
  const long long lbl = labels[n];
  const float p = qf[(size_t)n * DIM + d] * ke[(size_t)lbl * DIM + d];
  const float dot = block128_reduce(p);
  if (d == 0) tgt[n] = dot * kinv[lbl];
}

// ---------------- main fused GEMM + sum(exp) kernel ----------------
// grid: (V_CHUNKS, numRowTiles/2); block: 256 = 8 wave32.
// Each wave: A tiles for rows rt0,rt1 resident (64 VGPRs), double-buffered
// B tile stream, 8x v_wmma_f32_16x16x32_bf16 + 16x v_exp_f32 per iteration.
__global__ void __launch_bounds__(256)
gemm_lse_kernel(const v16bf* __restrict__ Apack,
                const v16bf* __restrict__ Bpack,
                float* __restrict__ sums,
                int numVTiles, int V, int N) {
  const int lane = threadIdx.x & 31;
  const int wave = threadIdx.x >> 5;
  const int rt0  = blockIdx.y * 2;
  const int rt1  = rt0 + 1;

  const v16bf* ap0 = Apack + (size_t)rt0 * 4 * 32;
  const v16bf* ap1 = Apack + (size_t)rt1 * 4 * 32;
  const v16bf a0 = ap0[0 * 32 + lane];
  const v16bf a1 = ap0[1 * 32 + lane];
  const v16bf a2 = ap0[2 * 32 + lane];
  const v16bf a3 = ap0[3 * 32 + lane];
  const v16bf a4 = ap1[0 * 32 + lane];
  const v16bf a5 = ap1[1 * 32 + lane];
  const v16bf a6 = ap1[2 * 32 + lane];
  const v16bf a7 = ap1[3 * 32 + lane];

  v8f acc0 = {};
  v8f acc1 = {};

  const int tilesPerChunk = (numVTiles + (int)gridDim.x - 1) / (int)gridDim.x;
  const int tBeg = blockIdx.x * tilesPerChunk;
  int tEnd = tBeg + tilesPerChunk;
  if (tEnd > numVTiles) tEnd = numVTiles;

  int t = tBeg + wave;
  v16bf b0, b1, b2, b3;
  if (t < tEnd) {
    const v16bf* bp = Bpack + (size_t)t * 4 * 32;
    b0 = bp[0 * 32 + lane];
    b1 = bp[1 * 32 + lane];
    b2 = bp[2 * 32 + lane];
    b3 = bp[3 * 32 + lane];
  }

  for (; t < tEnd; t += 8) {
    // issue next tile's loads first (clamped index -> branch-free pipeline)
    const int tn = (t + 8 < tEnd) ? (t + 8) : (tEnd - 1);
    const v16bf* np = Bpack + (size_t)tn * 4 * 32;
    const v16bf n0 = np[0 * 32 + lane];
    const v16bf n1 = np[1 * 32 + lane];
    const v16bf n2 = np[2 * 32 + lane];
    const v16bf n3 = np[3 * 32 + lane];

    v8f c0 = {};
    c0 = __builtin_amdgcn_wmma_f32_16x16x32_bf16(false, a0, false, b0, (short)0, c0, false, false);
    c0 = __builtin_amdgcn_wmma_f32_16x16x32_bf16(false, a1, false, b1, (short)0, c0, false, false);
    c0 = __builtin_amdgcn_wmma_f32_16x16x32_bf16(false, a2, false, b2, (short)0, c0, false, false);
    c0 = __builtin_amdgcn_wmma_f32_16x16x32_bf16(false, a3, false, b3, (short)0, c0, false, false);
    v8f c1 = {};
    c1 = __builtin_amdgcn_wmma_f32_16x16x32_bf16(false, a4, false, b0, (short)0, c1, false, false);
    c1 = __builtin_amdgcn_wmma_f32_16x16x32_bf16(false, a5, false, b1, (short)0, c1, false, false);
    c1 = __builtin_amdgcn_wmma_f32_16x16x32_bf16(false, a6, false, b2, (short)0, c1, false, false);
    c1 = __builtin_amdgcn_wmma_f32_16x16x32_bf16(false, a7, false, b3, (short)0, c1, false, false);

    // C layout: VGPR r, lane l -> row = r + 8*(l>=16), col = l&15
    const bool colOK = (t * 16 + (lane & 15)) < V;
    if (colOK) {
#pragma unroll
      for (int i = 0; i < 8; ++i) acc0[i] += __expf(c0[i]);  // |logit|<=1: safe
#pragma unroll
      for (int i = 0; i < 8; ++i) acc1[i] += __expf(c1[i]);
    }

    b0 = n0; b1 = n1; b2 = n2; b3 = n3;
  }

  // reduce the 16 lanes sharing each row, then merge across blocks
#pragma unroll
  for (int r = 0; r < 8; ++r) {
    float v = acc0[r];
    v += __shfl_xor(v, 1, 32);
    v += __shfl_xor(v, 2, 32);
    v += __shfl_xor(v, 4, 32);
    v += __shfl_xor(v, 8, 32);   // stays within each 16-lane half
    const int row = rt0 * 16 + r + ((lane >> 4) << 3);
    if ((lane & 15) == 0 && row < N) atomicAdd(&sums[row], v);
  }
#pragma unroll
  for (int r = 0; r < 8; ++r) {
    float v = acc1[r];
    v += __shfl_xor(v, 1, 32);
    v += __shfl_xor(v, 2, 32);
    v += __shfl_xor(v, 4, 32);
    v += __shfl_xor(v, 8, 32);
    const int row = rt1 * 16 + r + ((lane >> 4) << 3);
    if ((lane & 15) == 0 && row < N) atomicAdd(&sums[row], v);
  }
}

// ---------------- final mean( log(sumexp) - tgt ) ----------------
__global__ void final_kernel(const float* __restrict__ sums,
                             const float* __restrict__ tgt,
                             float* __restrict__ out, int N) {
  __shared__ float sm[256];
  float loc = 0.0f;
  for (int n = threadIdx.x; n < N; n += 256) loc += logf(sums[n]) - tgt[n];
  sm[threadIdx.x] = loc;
  __syncthreads();
#pragma unroll
  for (int off = 128; off > 0; off >>= 1) {
    if ((int)threadIdx.x < off) sm[threadIdx.x] += sm[threadIdx.x + off];
    __syncthreads();
  }
  if (threadIdx.x == 0) out[0] = sm[0] / (float)N;
}

// ---------------------------------------------------------------------------

extern "C" void kernel_launch(void* const* d_in, const int* in_sizes, int n_in,
                              void* d_out, int out_size, void* d_ws, size_t ws_size,
                              hipStream_t stream) {
  const float*     qe     = (const float*)d_in[0];
  const float*     ke     = (const float*)d_in[1];
  const long long* loc    = (const long long*)d_in[2];  // int64 (N,2)
  const long long* labels = (const long long*)d_in[3];  // int64 (N,)
  float* out = (float*)d_out;

  const int V = in_sizes[1] / DIM;            // 100000
  const int N = in_sizes[3];                  // 2048
  const int numVTiles = (V + 15) / 16;        // 6250
  const int numRTiles = (N + 15) / 16;        // 128
  const int rtPairs   = (numRTiles + 1) / 2;  // 64
  const int rtPadded  = rtPairs * 2;          // A tiles allocated (zero-padded)

  // carve workspace (256B aligned chunks)
  size_t off = 0;
  auto carve = [&](size_t bytes) -> void* {
    void* p = (char*)d_ws + off;
    off += (bytes + 255) & ~(size_t)255;
    return p;
  };
  __bf16* Bpack = (__bf16*)carve((size_t)numVTiles * 4 * 512 * sizeof(__bf16)); // ~25.6MB
  __bf16* Apack = (__bf16*)carve((size_t)rtPadded * 4 * 512 * sizeof(__bf16));
  float*  qf    = (float*)carve((size_t)N * DIM * sizeof(float));
  float*  kinv  = (float*)carve((size_t)V * sizeof(float));
  float*  sums  = (float*)carve((size_t)N * sizeof(float));
  float*  tgt   = (float*)carve((size_t)N * sizeof(float));

  zero_kernel<<<(N + 255) / 256, 256, 0, stream>>>(sums, N);
  gather_norm_q_kernel<<<N, DIM, 0, stream>>>(qe, loc, qf);
  knorm_kernel<<<V, DIM, 0, stream>>>(ke, kinv);
  pack_q_kernel<<<rtPadded * 4, 512, 0, stream>>>(qf, Apack, N);
  pack_k_kernel<<<numVTiles * 4, 512, 0, stream>>>(ke, kinv, Bpack, V);
  tgt_kernel<<<N, DIM, 0, stream>>>(qf, ke, kinv, labels, tgt);

  dim3 grid(32, rtPairs);   // 2048 blocks x 8 wave32
  gemm_lse_kernel<<<grid, 256, 0, stream>>>((const v16bf*)Apack, (const v16bf*)Bpack,
                                            sums, numVTiles, V, N);
  final_kernel<<<1, 256, 0, stream>>>(sums, tgt, out, N);
}
